// RelativeMultiheadAttention_61143154426157
// MI455X (gfx1250) — matrix operand order repaired and checked by reading, not compiled, not run on hardware
//
#include <hip/hip_runtime.h>
#include <hip/hip_bf16.h>

typedef __attribute__((ext_vector_type(16))) __bf16 v16bf_t;
typedef __attribute__((ext_vector_type(8)))  float  v8f_t;

// ---------------------------------------------------------------------------
// WMMA GEMM: C[M,N] = A[M,K] * B[K,N] (+ bias[N]), fp32 in/out, bf16x3 inside.
// Block: 256 threads (8 waves). Macro tile 128(M) x 64(N), K-step 32.
// Wave w -> 32x32 region: wm = (w&3)*32, wn = (w>>2)*32; 2x2 WMMA subtiles.
// ---------------------------------------------------------------------------
#define TILE_M 128
#define TILE_N 64
#define TILE_K 32
#define LDS_STRIDE 34   // 32 + 2 pad (17 dwords; coprime with 64 banks)

__global__ __launch_bounds__(256)
void wmma_gemm_bf16x3(const float* __restrict__ A, const float* __restrict__ B,
                      const float* __restrict__ bias, float* __restrict__ Cout,
                      int M, int N, int K) {
  __shared__ __bf16 Ah[TILE_M * LDS_STRIDE];
  __shared__ __bf16 Al[TILE_M * LDS_STRIDE];
  __shared__ __bf16 Bh[TILE_N * LDS_STRIDE];   // transposed: [n][k]
  __shared__ __bf16 Bl[TILE_N * LDS_STRIDE];

  const int tid  = threadIdx.x;
  const int m0   = blockIdx.y * TILE_M;
  const int n0   = blockIdx.x * TILE_N;
  const int wave = tid >> 5;
  const int lane = tid & 31;
  const int wm   = (wave & 3) << 5;   // 0,32,64,96
  const int wn   = (wave >> 2) << 5;  // 0,32
  const int lrow = lane & 15;
  const int half = lane >> 4;         // 0: lanes 0-15, 1: lanes 16-31
  const int kbA  = half * 16;         // A K-base (contiguous K, matches B)
  const int kbB  = half * 16;         // B K-base (contiguous K, matches A)

  v8f_t acc[2][2];
  #pragma unroll
  for (int i = 0; i < 2; ++i)
    #pragma unroll
    for (int j = 0; j < 2; ++j) {
      v8f_t z = {};
      acc[i][j] = z;
    }

  for (int k0 = 0; k0 < K; k0 += TILE_K) {
    // ---- stage A: 128x32 fp32 -> hi/lo bf16 in LDS (coalesced float4) ----
    #pragma unroll
    for (int j = 0; j < 4; ++j) {
      int idx = j * 256 + tid;
      int r   = idx >> 3;          // 0..127
      int cv  = (idx & 7) << 2;    // 0,4,...,28
      const float4 v = *(const float4*)(A + (size_t)(m0 + r) * K + k0 + cv);
      float xs[4] = {v.x, v.y, v.z, v.w};
      #pragma unroll
      for (int i = 0; i < 4; ++i) {
        float x = xs[i];
        __bf16 h = (__bf16)x;
        __bf16 l = (__bf16)(x - (float)h);
        Ah[r * LDS_STRIDE + cv + i] = h;
        Al[r * LDS_STRIDE + cv + i] = l;
      }
    }
    // ---- stage B: 32x64 fp32 -> hi/lo bf16, transposed [n][k] in LDS ----
    #pragma unroll
    for (int j = 0; j < 2; ++j) {
      int idx = j * 256 + tid;
      int r   = idx >> 4;          // k row 0..31
      int cv  = (idx & 15) << 2;   // n col 0,4,...,60
      const float4 v = *(const float4*)(B + (size_t)(k0 + r) * N + n0 + cv);
      float xs[4] = {v.x, v.y, v.z, v.w};
      #pragma unroll
      for (int i = 0; i < 4; ++i) {
        float x = xs[i];
        __bf16 h = (__bf16)x;
        __bf16 l = (__bf16)(x - (float)h);
        Bh[(cv + i) * LDS_STRIDE + r] = h;
        Bl[(cv + i) * LDS_STRIDE + r] = l;
      }
    }
    __syncthreads();

    // ---- gather fragments, contiguous K = 16*half + s for A and B ----
    v16bf_t ah[2], al[2], bh[2], bl[2];
    #pragma unroll
    for (int s = 0; s < 2; ++s) {
      const int mrow = wm + s * 16 + lrow;
      #pragma unroll
      for (int i = 0; i < 8; ++i) {
        // VGPR i: K = kbA + 2i
        int koff = (i << 1) + kbA;
        ah[s][2 * i]     = Ah[mrow * LDS_STRIDE + koff];
        ah[s][2 * i + 1] = Ah[mrow * LDS_STRIDE + koff + 1];
        al[s][2 * i]     = Al[mrow * LDS_STRIDE + koff];
        al[s][2 * i + 1] = Al[mrow * LDS_STRIDE + koff + 1];
      }
      const int ncol = wn + s * 16 + lrow;
      #pragma unroll
      for (int v = 0; v < 8; ++v) {
        // lanes 0-15: K=0..15, lanes 16-31: K=16..31; VGPR v -> K = kbB+2v
        bh[s][2 * v]     = Bh[ncol * LDS_STRIDE + kbB + 2 * v];
        bh[s][2 * v + 1] = Bh[ncol * LDS_STRIDE + kbB + 2 * v + 1];
        bl[s][2 * v]     = Bl[ncol * LDS_STRIDE + kbB + 2 * v];
        bl[s][2 * v + 1] = Bl[ncol * LDS_STRIDE + kbB + 2 * v + 1];
      }
    }

    // ---- bf16x3: hi*hi + lo*hi + hi*lo ----
    #pragma unroll
    for (int sm = 0; sm < 2; ++sm)
      #pragma unroll
      for (int sn = 0; sn < 2; ++sn) {
        acc[sm][sn] = __builtin_amdgcn_wmma_f32_16x16x32_bf16(
            false, ah[sm], false, bh[sn], (short)0, acc[sm][sn], false, false);
        acc[sm][sn] = __builtin_amdgcn_wmma_f32_16x16x32_bf16(
            false, al[sm], false, bh[sn], (short)0, acc[sm][sn], false, false);
        acc[sm][sn] = __builtin_amdgcn_wmma_f32_16x16x32_bf16(
            false, ah[sm], false, bl[sn], (short)0, acc[sm][sn], false, false);
      }
    __syncthreads();
  }

  // ---- store: C/D layout -> lane n = lane&15, VGPR r -> M = r + 8*half ----
  #pragma unroll
  for (int sm = 0; sm < 2; ++sm)
    #pragma unroll
    for (int sn = 0; sn < 2; ++sn)
      #pragma unroll
      for (int r = 0; r < 8; ++r) {
        int row = m0 + wm + sm * 16 + half * 8 + r;
        int col = n0 + wn + sn * 16 + lrow;
        float o = acc[sm][sn][r];
        if (bias) o += bias[col];
        Cout[(size_t)row * N + col] = o;
      }
}

// ---------------------------------------------------------------------------
// Y[b,e] = sum_s X[b,s,e]
// ---------------------------------------------------------------------------
__global__ void colsum_kernel(const float* __restrict__ X, float* __restrict__ Y,
                              int Bn, int S, int E) {
  int t = blockIdx.x * blockDim.x + threadIdx.x;
  if (t >= Bn * E) return;
  int b = t / E, e = t % E;
  const float* p = X + (size_t)b * S * E + e;
  float acc = 0.f;
  for (int s = 0; s < S; ++s) acc += p[(size_t)s * E];
  Y[t] = acc;
}

// ---------------------------------------------------------------------------
// Y[b,n] = X[b,:] @ W[:,n] + scale*bias[n]   (tiny M=2 GEMM)
// ---------------------------------------------------------------------------
__global__ void rowvec_gemm_kernel(const float* __restrict__ X, const float* __restrict__ W,
                                   const float* __restrict__ bias, float scale_bias,
                                   float* __restrict__ Y, int Bn, int E, int N) {
  int t = blockIdx.x * blockDim.x + threadIdx.x;
  if (t >= Bn * N) return;
  int b = t / N, n = t % N;
  float acc = bias ? scale_bias * bias[n] : 0.f;
  const float* x = X + (size_t)b * E;
  for (int k = 0; k < E; ++k) acc += x[k] * W[(size_t)k * N + n];
  Y[t] = acc;
}

// ---------------------------------------------------------------------------
// C[b,0,e]=0 ; C[b,m,e] = sum_{s<m} R[b,s,e]   (prefix of R-projection)
// ---------------------------------------------------------------------------
__global__ void prefix_kernel(const float* __restrict__ R, float* __restrict__ C,
                              int Bn, int S, int E) {
  int t = blockIdx.x * blockDim.x + threadIdx.x;
  if (t >= Bn * E) return;
  int b = t / E, e = t % E;
  const float* r = R + (size_t)b * S * E + e;
  float* c = C + (size_t)b * (S + 1) * E + e;
  float acc = 0.f;
  c[0] = 0.f;
  for (int s = 0; s < S; ++s) {
    acc += r[(size_t)s * E];
    c[(size_t)(s + 1) * E] = acc;
  }
}

// ---------------------------------------------------------------------------
// Per (b,q,h): row-sum of attention scores via factorization, then
// alpha[b,q,h*HD+d] = S_A * vsum[b,h*HD+d].
//   content: (q+U) . ksum
//   shifted pos: (q+V)[q] . (C[S]-C[S-1-q])  +  (q+V)[q+1] . C[S-2-q]
// ---------------------------------------------------------------------------
__global__ void score_alpha_kernel(const float* __restrict__ qp, const float* __restrict__ U,
                                   const float* __restrict__ V, const float* __restrict__ ksum,
                                   const float* __restrict__ Cpre, const float* __restrict__ vsum,
                                   float* __restrict__ alpha, int Bn, int S, int H, int HD) {
  int t = blockIdx.x * blockDim.x + threadIdx.x;
  if (t >= Bn * S * H) return;
  int h = t % H;
  int q = (t / H) % S;
  int b = t / (H * S);
  const int E = H * HD;

  const float* qrow = qp + ((size_t)b * S + q) * E + h * HD;
  const float* Uh   = U + h * HD;
  const float* Vh   = V + h * HD;
  const float* ks   = ksum + (size_t)b * E + h * HD;
  const float* ctop = Cpre + ((size_t)b * (S + 1) + S) * E + h * HD;
  const float* c1   = Cpre + ((size_t)b * (S + 1) + (S - 1 - q)) * E + h * HD;

  float dotc = 0.f, dot1 = 0.f, dot2 = 0.f;
  for (int d = 0; d < HD; ++d) {
    float qv = qrow[d];
    dotc += (qv + (float)Uh[d]) * ks[d];
    dot1 += (qv + (float)Vh[d]) * (ctop[d] - c1[d]);
  }
  if (q < S - 1) {
    const float* qnext = qrow + E;
    const float* c2 = Cpre + ((size_t)b * (S + 1) + (S - 2 - q)) * E + h * HD;
    for (int d = 0; d < HD; ++d) dot2 += (qnext[d] + Vh[d]) * c2[d];
  }
  float sA = (dotc + dot1 + dot2) * (1.0f / 32.0f);  // / sqrt(E), sqrt(1024)=32

  const float* vs = vsum + (size_t)b * E + h * HD;
  float* out = alpha + ((size_t)b * S + q) * E + h * HD;
  for (int d = 0; d < HD; ++d) out[d] = sA * vs[d];
}

// ---------------------------------------------------------------------------
extern "C" void kernel_launch(void* const* d_in, const int* in_sizes, int n_in,
                              void* d_out, int out_size, void* d_ws, size_t ws_size,
                              hipStream_t stream) {
  const float* query = (const float*)d_in[0];
  const float* key   = (const float*)d_in[1];
  const float* value = (const float*)d_in[2];
  const float* pos   = (const float*)d_in[3];
  const float* Wq    = (const float*)d_in[4];
  const float* bq    = (const float*)d_in[5];
  const float* Wk    = (const float*)d_in[6];
  const float* bk    = (const float*)d_in[7];
  const float* Wv    = (const float*)d_in[8];
  const float* bv    = (const float*)d_in[9];
  const float* Wp    = (const float*)d_in[10];
  const float* U     = (const float*)d_in[11];
  const float* V     = (const float*)d_in[12];
  const float* Wo    = (const float*)d_in[13];
  const float* bo    = (const float*)d_in[14];
  float* out = (float*)d_out;

  const int B = 2, S = 1024, E = 1024, H = 16, HD = 64;

  float* ws    = (float*)d_ws;
  float* qproj = ws;  ws += (size_t)B * S * E;        // 8 MB
  float* Rproj = ws;  ws += (size_t)B * S * E;        // 8 MB
  float* Cbuf  = ws;  ws += (size_t)B * (S + 1) * E;  // 8.4 MB
  float* alpha = ws;  ws += (size_t)B * S * E;        // 8 MB
  float* ksum  = ws;  ws += B * E;
  float* vsum  = ws;  ws += B * E;
  float* keyin = ws;  ws += B * E;
  float* valin = ws;  ws += B * E;

  // k/v projections collapse: ksum = (sum_s key) @ Wk + S*bk (ditto value)
  colsum_kernel<<<(B * E + 255) / 256, 256, 0, stream>>>(key, keyin, B, S, E);
  colsum_kernel<<<(B * E + 255) / 256, 256, 0, stream>>>(value, valin, B, S, E);
  rowvec_gemm_kernel<<<(B * E + 255) / 256, 256, 0, stream>>>(keyin, Wk, bk, (float)S, ksum, B, E, E);
  rowvec_gemm_kernel<<<(B * E + 255) / 256, 256, 0, stream>>>(valin, Wv, bv, (float)S, vsum, B, E, E);

  dim3 ggrid(E / TILE_N, (B * S) / TILE_M);  // (16, 16)
  wmma_gemm_bf16x3<<<ggrid, 256, 0, stream>>>(query, Wq, bq, qproj, B * S, E, E);
  wmma_gemm_bf16x3<<<ggrid, 256, 0, stream>>>(pos, Wp, nullptr, Rproj, B * S, E, E);

  prefix_kernel<<<(B * E + 255) / 256, 256, 0, stream>>>(Rproj, Cbuf, B, S, E);
  score_alpha_kernel<<<(B * S * H + 255) / 256, 256, 0, stream>>>(
      qproj, U, V, ksum, Cbuf, vsum, alpha, B, S, H, HD);

  wmma_gemm_bf16x3<<<ggrid, 256, 0, stream>>>(alpha, Wo, bo, out, B * S, E, E);
}
